// BailingMoEForCausalLM_47553877901443
// MI455X (gfx1250) — matrix operand order repaired and checked by reading, not compiled, not run on hardware
//
#include <hip/hip_runtime.h>

// ---------------- problem constants (match reference) ----------------
#define Bsz   2
#define Ssz   1024
#define Hdim  2048
#define HQn   16
#define HKVn  4
#define Dh    128
#define En    32
#define Kn    4
#define CAPn  512
#define IMn   512
#define SIMn  1024
#define Tn    (Bsz * Ssz)
#define EPSf  1e-6f

typedef unsigned short bf16u;
typedef __attribute__((ext_vector_type(16))) __bf16 v16bf;
typedef __attribute__((ext_vector_type(2)))  __bf16 v2bf;
typedef __attribute__((ext_vector_type(8)))  float  v8f;

union Frag16 { int4 q[2]; v16bf v; };

// native bf16 convert (backend selects v_cvt*_bf16_f32)
__device__ __forceinline__ bf16u f2bf(float f) {
    union { __bf16 h; bf16u u; } cv;
    cv.h = (__bf16)f;
    return cv.u;
}
// two floats -> one v_cvt_pk_bf16_f32 (build_vector of two fptruncs)
__device__ __forceinline__ unsigned pk2bf(float a, float b) {
    union { v2bf v; unsigned u; } cv;
    cv.v.x = (__bf16)a;
    cv.v.y = (__bf16)b;
    return cv.u;
}
__device__ __forceinline__ float bf2f(bf16u b) {
    return __uint_as_float(((unsigned int)b) << 16);
}

// gfx1250 async global->LDS copy (16B per lane), tracked by ASYNCcnt
__device__ __forceinline__ void async_g2l_b128(unsigned lds_addr, unsigned long long gaddr) {
    asm volatile("global_load_async_to_lds_b128 %0, %1, off"
                 :: "v"(lds_addr), "v"(gaddr) : "memory");
}
__device__ __forceinline__ void wait_async0() {
    asm volatile("s_wait_asynccnt 0x0" ::: "memory");
}

// =====================================================================
// RMSNorm (fp32 in, bf16 out), one block per row of H
// =====================================================================
__launch_bounds__(256)
__global__ void k_rmsnorm(const float* __restrict__ x, const float* __restrict__ w,
                          bf16u* __restrict__ obf) {
    const int t = blockIdx.x;
    const float* xr = x + (long long)t * Hdim;
    __shared__ float red[8];
    float s = 0.f;
    for (int i = threadIdx.x; i < Hdim; i += 256) { float v = xr[i]; s += v * v; }
    for (int m = 16; m >= 1; m >>= 1) s += __shfl_xor(s, m, 32);
    if ((threadIdx.x & 31) == 0) red[threadIdx.x >> 5] = s;
    __syncthreads();
    if (threadIdx.x < 8) {
        float v = red[threadIdx.x];
        for (int m = 4; m >= 1; m >>= 1) v += __shfl_xor(v, m, 8);
        if (threadIdx.x == 0) red[0] = v;
    }
    __syncthreads();
    const float inv = rsqrtf(red[0] / (float)Hdim + EPSf);
    for (int i = threadIdx.x; i < Hdim; i += 256)
        obf[(long long)t * Hdim + i] = f2bf(xr[i] * inv * w[i]);
}

// =====================================================================
// WMMA GEMM: C[M,N] = A[M,K](bf16) x B[K,N](fp32->bf16) (+R)
// 128x64 block tile, 4 waves (each 32x64), K-step 32, double-buffered:
//   A tile : global_load_async_to_lds_b128 (no VGPR round trip)
//   B tile : pipelined fp32 loads -> v_cvt_pk_bf16_f32 -> b32 LDS stores
// =====================================================================
#define LDT 40   // LDS row stride in bf16 elems (80B, 16B-aligned)

__launch_bounds__(128)
__global__ void k_gemm(const bf16u* __restrict__ A, const float* __restrict__ Bm,
                       float* __restrict__ Cf, bf16u* __restrict__ Cbf,
                       const float* __restrict__ R,
                       int M, int N, int Kd,
                       long long sA, long long sB, long long sC, long long sR) {
    __shared__ bf16u As[2][128 * LDT];
    __shared__ bf16u Bt[2][64 * LDT];   // B tile transposed: Bt[n][k]

    const int e = blockIdx.z;
    const bf16u* Ab = A + (long long)e * sA;
    const float* Bb = Bm + (long long)e * sB;
    const long long cbase = (long long)e * sC;
    const int m0 = blockIdx.y * 128, n0 = blockIdx.x * 64;
    const int tid = threadIdx.x;
    const int wave = tid >> 5, lane = tid & 31;
    const int half = lane >> 4, lr = lane & 15;

    v8f acc[2][4];
    #pragma unroll
    for (int mi = 0; mi < 2; mi++)
        #pragma unroll
        for (int n = 0; n < 4; n++)
            #pragma unroll
            for (int r = 0; r < 8; r++) acc[mi][n][r] = 0.f;

    // ---- A tile stage: 128x32 bf16 = 512 x 16B chunks, 4 per thread ----
    auto stageA = [&](int buf, int kt) {
        #pragma unroll
        for (int it = 0; it < 4; it++) {
            int c = tid + it * 128;
            int row = c >> 2, col8 = (c & 3) << 3;
            unsigned lo = (unsigned)(unsigned long long)&As[buf][row * LDT + col8];
            unsigned long long ga =
                (unsigned long long)(Ab + (long long)(m0 + row) * Kd + kt + col8);
            async_g2l_b128(lo, ga);
        }
    };
    // ---- B tile: 32x64 fp32; each thread owns 2 chunks of (2k x 4n) ----
    auto loadB = [&](int kt, float4 br[2][2]) {
        #pragma unroll
        for (int it = 0; it < 2; it++) {
            int c = tid + it * 128;
            int k = (c & 15) << 1, n = (c >> 4) << 2;
            const float* bp = Bb + (long long)(kt + k) * N + n0 + n;
            br[it][0] = *(const float4*)bp;        // row k
            br[it][1] = *(const float4*)(bp + N);  // row k+1
        }
    };
    auto storeB = [&](int buf, float4 br[2][2]) {
        #pragma unroll
        for (int it = 0; it < 2; it++) {
            int c = tid + it * 128;
            int k = (c & 15) << 1, n = (c >> 4) << 2;
            const float* r0 = (const float*)&br[it][0];
            const float* r1 = (const float*)&br[it][1];
            #pragma unroll
            for (int j = 0; j < 4; j++)            // pack {k,k+1} -> one b32
                *(unsigned*)(&Bt[buf][(n + j) * LDT + k]) = pk2bf(r0[j], r1[j]);
        }
    };

    // prologue: stage tile 0 into buffer 0
    {
        float4 br[2][2];
        stageA(0, 0);
        loadB(0, br);
        storeB(0, br);
        wait_async0();
    }
    __syncthreads();

    int cur = 0;
    for (int kt = 0; kt < Kd; kt += 32) {
        const int nxt = cur ^ 1;
        const bool hasNext = (kt + 32) < Kd;
        float4 br[2][2];
        if (hasNext) {            // issue next tile's traffic before computing
            stageA(nxt, kt + 32);
            loadB(kt + 32, br);
        }

        // B fragments (shared by both M sub-tiles of this wave)
        Frag16 bf4[4];
        #pragma unroll
        for (int n = 0; n < 4; n++) {
            const bf16u* brr = &Bt[cur][(n * 16 + lr) * LDT + half * 16];
            bf4[n].q[0] = *(const int4*)(brr);
            bf4[n].q[1] = *(const int4*)(brr + 8);
        }
        #pragma unroll
        for (int mi = 0; mi < 2; mi++) {
            Frag16 af;
            const bf16u* ar = &As[cur][(wave * 32 + mi * 16 + lr) * LDT];
            af.q[0] = *(const int4*)(ar + half * 16);
            af.q[1] = *(const int4*)(ar + half * 16 + 8);
            #pragma unroll
            for (int n = 0; n < 4; n++)
                acc[mi][n] = __builtin_amdgcn_wmma_f32_16x16x32_bf16(
                    false, af.v, false, bf4[n].v, (short)0, acc[mi][n], false, false);
        }

        if (hasNext) {
            storeB(nxt, br);
            wait_async0();        // A async copies into nxt finished
        }
        __syncthreads();
        cur = nxt;
    }

    // epilogue: optional residual, fp32 and/or bf16 stores
    #pragma unroll
    for (int mi = 0; mi < 2; mi++) {
        #pragma unroll
        for (int n = 0; n < 4; n++) {
            #pragma unroll
            for (int r = 0; r < 8; r++) {
                int row = m0 + wave * 32 + mi * 16 + r + half * 8;
                int col = n0 + n * 16 + lr;
                long long off = cbase + (long long)row * N + col;
                float v = acc[mi][n][r];
                if (R)   v += R[(long long)blockIdx.z * sR + (long long)row * N + col];
                if (Cf)  Cf[off] = v;
                if (Cbf) Cbf[off] = f2bf(v);
            }
        }
    }
}

// =====================================================================
// per-head qk RMSNorm + neox RoPE  (fp32 in, bf16 out), block = D = 128
// =====================================================================
__launch_bounds__(128)
__global__ void k_qknorm_rope(const float* __restrict__ X, const float* __restrict__ wn,
                              const int* __restrict__ pos, bf16u* __restrict__ O, int NH) {
    const int th = blockIdx.x;          // t*NH + h
    const int t  = th / NH;
    const long long base = (long long)th * Dh;
    const int i = threadIdx.x;          // 0..127
    float v = X[base + i];
    __shared__ float red[4];
    __shared__ float sh[Dh];
    float s = v * v;
    for (int m = 16; m >= 1; m >>= 1) s += __shfl_xor(s, m, 32);
    if ((i & 31) == 0) red[i >> 5] = s;
    __syncthreads();
    const float tot = red[0] + red[1] + red[2] + red[3];
    const float rn = v * rsqrtf(tot / (float)Dh + EPSf) * wn[i];
    sh[i] = rn;
    __syncthreads();
    const float p = (float)pos[t];
    float out;
    if (i < Dh / 2) {
        float fr = p * __powf(1.0e6f, -((float)(2 * i)) / (float)Dh);
        out = rn * __cosf(fr) - sh[i + Dh / 2] * __sinf(fr);
    } else {
        int j = i - Dh / 2;
        float fr = p * __powf(1.0e6f, -((float)(2 * j)) / (float)Dh);
        out = rn * __cosf(fr) + sh[j] * __sinf(fr);
    }
    O[base + i] = f2bf(out);
}

// =====================================================================
// V fp32 [T,HKV,D]  ->  bf16 transposed [B,HKV,D,S] for PV B-fragments
// =====================================================================
__launch_bounds__(256)
__global__ void k_vt(const float* __restrict__ vf, bf16u* __restrict__ vt) {
    const int t = blockIdx.x;
    const int b = t / Ssz, s = t % Ssz;
    for (int i = threadIdx.x; i < HKVn * Dh; i += 256) {
        int hk = i / Dh, d = i % Dh;
        vt[(((long long)(b * HKVn + hk)) * Dh + d) * Ssz + s] =
            f2bf(vf[(long long)t * HKVn * Dh + i]);
    }
}

// =====================================================================
// Flash attention, causal, GQA. 1 wave per 16-query tile per head.
// QK^T and P*V via v_wmma_f32_16x16x32_bf16, fp32 online softmax.
// =====================================================================
__launch_bounds__(32)
__global__ void k_attn(const bf16u* __restrict__ Qb, const bf16u* __restrict__ Kb,
                       const bf16u* __restrict__ VT, bf16u* __restrict__ Ob) {
    const int qt = blockIdx.x, h = blockIdx.y, b = blockIdx.z;
    const int hk = h / (HQn / HKVn);
    const int lane = threadIdx.x, half = lane >> 4, lr = lane & 15;
    __shared__ bf16u Pl[16 * 32];

    // Q A-fragments over 4 D-chunks of 32
    Frag16 qf[4];
    {
        const bf16u* qrow = Qb + ((long long)(b * Ssz + qt * 16 + lr) * HQn + h) * Dh;
        #pragma unroll
        for (int c = 0; c < 4; c++) {
            qf[c].q[0] = *(const int4*)(qrow + c * 32 + half * 16);
            qf[c].q[1] = *(const int4*)(qrow + c * 32 + half * 16 + 8);
        }
    }

    v8f o[8];
    float mrow[8], lrow[8];
    #pragma unroll
    for (int n = 0; n < 8; n++)
        #pragma unroll
        for (int r = 0; r < 8; r++) o[n][r] = 0.f;
    #pragma unroll
    for (int r = 0; r < 8; r++) { mrow[r] = -1e30f; lrow[r] = 0.f; }

    const int qmax = qt * 16 + 15;
    const float scale = 0.088388347648318447f;   // 1/sqrt(128)

    for (int j0 = 0; j0 <= qmax; j0 += 32) {
        v8f sc[2];
        #pragma unroll
        for (int sub = 0; sub < 2; sub++) {
            const int kbase = j0 + sub * 16;
            if (kbase > qmax) {                   // uniform branch: tile fully masked
                #pragma unroll
                for (int r = 0; r < 8; r++) sc[sub][r] = -1e30f;
                continue;
            }
            #pragma unroll
            for (int r = 0; r < 8; r++) sc[sub][r] = 0.f;
            const bf16u* krow = Kb + ((long long)(b * Ssz + kbase + lr) * HKVn + hk) * Dh;
            #pragma unroll
            for (int c = 0; c < 4; c++) {
                Frag16 kf;                        // B-frag: col=key, K=d
                kf.q[0] = *(const int4*)(krow + c * 32 + half * 16);
                kf.q[1] = *(const int4*)(krow + c * 32 + half * 16 + 8);
                sc[sub] = __builtin_amdgcn_wmma_f32_16x16x32_bf16(
                    false, qf[c].v, false, kf.v, (short)0, sc[sub], false, false);
            }
            #pragma unroll
            for (int r = 0; r < 8; r++) {
                int qi = qt * 16 + r + half * 8;
                int ki = kbase + lr;
                sc[sub][r] = (ki <= qi) ? sc[sub][r] * scale : -1e30f;
            }
        }

        // online softmax over the 32-key slab (rows striped across 16-lane halves)
        #pragma unroll
        for (int r = 0; r < 8; r++) {
            float mx = fmaxf(sc[0][r], sc[1][r]);
            for (int m = 8; m >= 1; m >>= 1) mx = fmaxf(mx, __shfl_xor(mx, m, 16));
            float mnew = fmaxf(mrow[r], mx);
            float p0 = __expf(sc[0][r] - mnew);
            float p1 = __expf(sc[1][r] - mnew);
            Pl[(r + half * 8) * 32 + lr]      = f2bf(p0);
            Pl[(r + half * 8) * 32 + 16 + lr] = f2bf(p1);
            float ps = p0 + p1;
            for (int m = 8; m >= 1; m >>= 1) ps += __shfl_xor(ps, m, 16);
            float corr = __expf(mrow[r] - mnew);
            lrow[r] = lrow[r] * corr + ps;
            mrow[r] = mnew;
            #pragma unroll
            for (int n = 0; n < 8; n++) o[n][r] *= corr;
        }
        __syncthreads();
        Frag16 pf;                                // P as A-fragment (16x32)
        pf.q[0] = *(const int4*)(&Pl[lr * 32 + half * 16]);
        pf.q[1] = *(const int4*)(&Pl[lr * 32 + half * 16 + 8]);
        __syncthreads();

        const bf16u* vb = VT + ((long long)(b * HKVn + hk) * Dh) * Ssz;
        #pragma unroll
        for (int n = 0; n < 8; n++) {             // B-frag: col=d, K=key
            const bf16u* vr = vb + (long long)(n * 16 + lr) * Ssz + j0 + half * 16;
            Frag16 vf;
            vf.q[0] = *(const int4*)(vr);
            vf.q[1] = *(const int4*)(vr + 8);
            o[n] = __builtin_amdgcn_wmma_f32_16x16x32_bf16(
                false, pf.v, false, vf.v, (short)0, o[n], false, false);
        }
    }

    const long long obase = ((long long)(b * Ssz + qt * 16) * HQn + h) * Dh;
    #pragma unroll
    for (int r = 0; r < 8; r++) {
        float inv = 1.0f / lrow[r];
        #pragma unroll
        for (int n = 0; n < 8; n++)
            Ob[obase + (long long)(r + half * 8) * HQn * Dh + n * 16 + lr] =
                f2bf(o[n][r] * inv);
    }
}

// =====================================================================
// sigmoid router + biased top-4, renormalized weights
// =====================================================================
__launch_bounds__(128)
__global__ void k_router(const bf16u* __restrict__ hbf, const float* __restrict__ gw,
                         const float* __restrict__ ebias, int* __restrict__ oidx,
                         float* __restrict__ ow) {
    const int t = blockIdx.x;
    const int tid = threadIdx.x, e = tid & 31, g = tid >> 5;
    __shared__ float part[4][En];
    __shared__ float sscore[En];
    __shared__ float sbias[En];
    float s = 0.f;
    const int span = Hdim / 4;
    for (int i = g * span; i < (g + 1) * span; i++)
        s += bf2f(hbf[(long long)t * Hdim + i]) * gw[(long long)i * En + e];
    part[g][e] = s;
    __syncthreads();
    if (tid < En) {
        float sc = part[0][e] + part[1][e] + part[2][e] + part[3][e];
        sc = 1.f / (1.f + __expf(-sc));
        sscore[e] = sc;
        sbias[e] = sc + ebias[e];
    }
    __syncthreads();
    if (tid == 0) {
        int sel[Kn]; float w[Kn]; float wsum = 0.f;
        for (int kk = 0; kk < Kn; kk++) {
            int best = 0; float bv = -1e30f;
            for (int e2 = 0; e2 < En; e2++) {
                bool taken = false;
                for (int j = 0; j < kk; j++) taken |= (sel[j] == e2);
                if (!taken && sbias[e2] > bv) { bv = sbias[e2]; best = e2; }
            }
            sel[kk] = best; w[kk] = sscore[best]; wsum += w[kk];
        }
        for (int kk = 0; kk < Kn; kk++) {
            oidx[t * Kn + kk] = sel[kk];
            ow[t * Kn + kk]   = w[kk] / wsum;
        }
    }
}

// =====================================================================
// dispatch: atomic slot alloc + bf16 scatter of token row into xbuf
// =====================================================================
__launch_bounds__(256)
__global__ void k_dispatch(const int* __restrict__ oidx, const bf16u* __restrict__ hbf,
                           int* __restrict__ counts, int* __restrict__ slot,
                           bf16u* __restrict__ xbuf) {
    const int id = blockIdx.x;            // t*K + k
    const int t = id >> 2;
    const int e = oidx[id];
    __shared__ int sp;
    if (threadIdx.x == 0) { sp = atomicAdd(&counts[e], 1); slot[id] = sp; }
    __syncthreads();
    const int p = sp;
    if (p >= CAPn) return;
    const int4* src = (const int4*)(hbf + (long long)t * Hdim);
    int4* dst = (int4*)(xbuf + ((long long)e * CAPn + p) * Hdim);
    dst[threadIdx.x] = src[threadIdx.x];  // 2048 bf16 = 256 x int4
}

// =====================================================================
// silu(g) * u  -> bf16
// =====================================================================
__launch_bounds__(256)
__global__ void k_silu_mul(const float* __restrict__ g, const float* __restrict__ u,
                           bf16u* __restrict__ o, long long n) {
    for (long long i = (long long)blockIdx.x * 256 + threadIdx.x; i < n;
         i += (long long)gridDim.x * 256) {
        float x = g[i];
        o[i] = f2bf((x / (1.f + __expf(-x))) * u[i]);
    }
}

// =====================================================================
// combine: out[t,:] += w * obuf[e, slot, :]
// =====================================================================
__launch_bounds__(256)
__global__ void k_combine(const int* __restrict__ oidx, const int* __restrict__ slot,
                          const float* __restrict__ ow, const bf16u* __restrict__ obuf,
                          float* __restrict__ out) {
    const int id = blockIdx.x;
    const int t = id >> 2;
    const int p = slot[id];
    if (p >= CAPn) return;
    const int e = oidx[id];
    const float w = ow[id];
    const bf16u* src = obuf + ((long long)e * CAPn + p) * Hdim;
    float* dst = out + (long long)t * Hdim;
    for (int i = threadIdx.x; i < Hdim; i += 256)
        atomicAdd(&dst[i], w * bf2f(src[i]));
}

// =====================================================================
// host side
// =====================================================================
extern "C" void kernel_launch(void* const* d_in, const int* in_sizes, int n_in,
                              void* d_out, int out_size, void* d_ws, size_t ws_size,
                              hipStream_t stream) {
    (void)in_sizes; (void)n_in; (void)out_size; (void)ws_size;
    const float* x     = (const float*)d_in[0];
    const int*   pos   = (const int*)  d_in[1];
    const float* ln1w  = (const float*)d_in[2];
    const float* ln2w  = (const float*)d_in[3];
    const float* wq    = (const float*)d_in[4];
    const float* wk    = (const float*)d_in[5];
    const float* wv    = (const float*)d_in[6];
    const float* wo    = (const float*)d_in[7];
    const float* qnw   = (const float*)d_in[8];
    const float* knw   = (const float*)d_in[9];
    const float* gatew = (const float*)d_in[10];
    const float* ebias = (const float*)d_in[11];
    const float* wge   = (const float*)d_in[12];
    const float* wue   = (const float*)d_in[13];
    const float* wde   = (const float*)d_in[14];
    const float* wsg   = (const float*)d_in[15];
    const float* wsu   = (const float*)d_in[16];
    const float* wsd   = (const float*)d_in[17];
    float* out = (float*)d_out;

    // ---- bump allocator over d_ws ----
    char* base = (char*)d_ws; size_t off = 0;
    auto alloc = [&](size_t bytes) -> void* {
        void* p = base + off;
        off = (off + bytes + 255) & ~(size_t)255;
        return p;
    };
    bf16u* h1_bf   = (bf16u*)alloc((size_t)Tn * Hdim * 2);
    float* q_f     = (float*)alloc((size_t)Tn * HQn * Dh * 4);
    float* k_f     = (float*)alloc((size_t)Tn * HKVn * Dh * 4);
    float* v_f     = (float*)alloc((size_t)Tn * HKVn * Dh * 4);
    bf16u* q_bf    = (bf16u*)alloc((size_t)Tn * HQn * Dh * 2);
    bf16u* k_bf    = (bf16u*)alloc((size_t)Tn * HKVn * Dh * 2);
    bf16u* vt_bf   = (bf16u*)alloc((size_t)Bsz * HKVn * Dh * Ssz * 2);
    bf16u* ao_bf   = (bf16u*)alloc((size_t)Tn * HQn * Dh * 2);
    float* h2_f    = (float*)alloc((size_t)Tn * Hdim * 4);
    bf16u* h2n_bf  = (bf16u*)alloc((size_t)Tn * Hdim * 2);
    float* sg_f    = (float*)alloc((size_t)Tn * SIMn * 4);
    float* su_f    = (float*)alloc((size_t)Tn * SIMn * 4);
    bf16u* sint_bf = (bf16u*)alloc((size_t)Tn * SIMn * 2);
    int*   oidx    = (int*)  alloc((size_t)Tn * Kn * 4);
    float* ow      = (float*)alloc((size_t)Tn * Kn * 4);
    int*   slot    = (int*)  alloc((size_t)Tn * Kn * 4);
    int*   counts  = (int*)  alloc((size_t)En * 4);
    bf16u* xbuf    = (bf16u*)alloc((size_t)En * CAPn * Hdim * 2);
    float* eg_f    = (float*)alloc((size_t)En * CAPn * IMn * 4);
    float* eu_f    = (float*)alloc((size_t)En * CAPn * IMn * 4);
    bf16u* eint_bf = (bf16u*)alloc((size_t)En * CAPn * IMn * 2);
    bf16u* obuf_bf = (bf16u*)alloc((size_t)En * CAPn * Hdim * 2);

    auto gemm = [&](const bf16u* A, const float* Bm, float* Cf, bf16u* Cbf,
                    const float* R, int M, int N, int Kd, int batch,
                    long long sA, long long sB, long long sC, long long sR) {
        dim3 g(N / 64, M / 128, batch);
        k_gemm<<<g, dim3(128), 0, stream>>>(A, Bm, Cf, Cbf, R, M, N, Kd, sA, sB, sC, sR);
    };

    // ---- attention block ----
    k_rmsnorm<<<Tn, 256, 0, stream>>>(x, ln1w, h1_bf);
    gemm(h1_bf, wq, q_f, nullptr, nullptr, Tn, HQn * Dh,  Hdim, 1, 0, 0, 0, 0);
    gemm(h1_bf, wk, k_f, nullptr, nullptr, Tn, HKVn * Dh, Hdim, 1, 0, 0, 0, 0);
    gemm(h1_bf, wv, v_f, nullptr, nullptr, Tn, HKVn * Dh, Hdim, 1, 0, 0, 0, 0);
    k_qknorm_rope<<<Tn * HQn,  128, 0, stream>>>(q_f, qnw, pos, q_bf, HQn);
    k_qknorm_rope<<<Tn * HKVn, 128, 0, stream>>>(k_f, knw, pos, k_bf, HKVn);
    k_vt<<<Tn, 256, 0, stream>>>(v_f, vt_bf);
    k_attn<<<dim3(Ssz / 16, HQn, Bsz), 32, 0, stream>>>(q_bf, k_bf, vt_bf, ao_bf);
    gemm(ao_bf, wo, h2_f, nullptr, x, Tn, Hdim, HQn * Dh, 1, 0, 0, 0, 0);

    // ---- ln2 + shared experts (down-proj fuses residual into d_out) ----
    k_rmsnorm<<<Tn, 256, 0, stream>>>(h2_f, ln2w, h2n_bf);
    gemm(h2n_bf, wsg, sg_f, nullptr, nullptr, Tn, SIMn, Hdim, 1, 0, 0, 0, 0);
    gemm(h2n_bf, wsu, su_f, nullptr, nullptr, Tn, SIMn, Hdim, 1, 0, 0, 0, 0);
    k_silu_mul<<<2048, 256, 0, stream>>>(sg_f, su_f, sint_bf, (long long)Tn * SIMn);
    gemm(sint_bf, wsd, out, nullptr, h2_f, Tn, Hdim, SIMn, 1, 0, 0, 0, 0);

    // ---- router + dispatch ----
    k_router<<<Tn, 128, 0, stream>>>(h2n_bf, gatew, ebias, oidx, ow);
    hipMemsetAsync(counts, 0, En * sizeof(int), stream);
    k_dispatch<<<Tn * Kn, 256, 0, stream>>>(oidx, h2n_bf, counts, slot, xbuf);

    // ---- expert grouped GEMMs (batched over grid.z) ----
    gemm(xbuf, wge, eg_f, nullptr, nullptr, CAPn, IMn, Hdim, En,
         (long long)CAPn * Hdim, (long long)Hdim * IMn, (long long)CAPn * IMn, 0);
    gemm(xbuf, wue, eu_f, nullptr, nullptr, CAPn, IMn, Hdim, En,
         (long long)CAPn * Hdim, (long long)Hdim * IMn, (long long)CAPn * IMn, 0);
    k_silu_mul<<<4096, 256, 0, stream>>>(eg_f, eu_f, eint_bf, (long long)En * CAPn * IMn);
    gemm(eint_bf, wde, nullptr, obuf_bf, nullptr, CAPn, Hdim, IMn, En,
         (long long)CAPn * IMn, (long long)IMn * Hdim, (long long)CAPn * Hdim, 0);

    // ---- weighted combine into output ----
    k_combine<<<Tn * Kn, 256, 0, stream>>>(oidx, slot, ow, obuf_bf, out);
}
